// LSTM_8873402434031
// MI455X (gfx1250) — 8-bit matrix operand order repaired and checked by reading, compiled, not run on hardware
//
#include <hip/hip_runtime.h>

#define BATCH  4096
#define TSTEPS 2048
#define HID    128
#define GATES  512
#define NOUT   8
#define MTILE  16
#define WPAD   144

typedef int          v8i   __attribute__((ext_vector_type(8)));
typedef float        v8f   __attribute__((ext_vector_type(8)));
typedef unsigned int u32x4 __attribute__((ext_vector_type(4)));

#define OFF_WHH1 0
#define OFF_WIH2 (512*WPAD)
#define OFF_WHH2 (2*512*WPAD)
#define OFF_W1   (3*512*WPAD)
#define OFF_W2   (OFF_W1 + 16*WPAD)
#define OFF_H1   (OFF_W2 + 16*WPAD)
#define OFF_H2   (OFF_H1 + 2*16*WPAD)
#define OFF_BS1  (OFF_H2 + 2*16*WPAD)
#define OFF_BS2  (OFF_BS1 + 512*4)
#define OFF_WX1  (OFF_BS2 + 512*4)
#define OFF_B1C  (OFF_WX1 + 512*4)
#define OFF_B2C  (OFF_B1C + 64)
#define OFF_X    (OFF_B2C + 64)
#define SMEM_SZ  (OFF_X + 128)

__device__ __forceinline__ float fast_rcp(float x) {
#if __has_builtin(__builtin_amdgcn_rcpf)
  return __builtin_amdgcn_rcpf(x);
#else
  return 1.0f / x;
#endif
}

__device__ __forceinline__ float fast_tanh(float x) {
#if __has_builtin(__builtin_amdgcn_tanhf)
  return __builtin_amdgcn_tanhf(x);
#elif __has_builtin(__builtin_amdgcn_tanh_f32)
  return __builtin_amdgcn_tanh_f32(x);
#else
  float xc = fminf(fmaxf(x, -15.0f), 15.0f);
  float e  = __expf(2.0f * xc);
  return (e - 1.0f) * fast_rcp(e + 1.0f);
#endif
}

__device__ __forceinline__ float fast_sigm(float x) {
#if __has_builtin(__builtin_amdgcn_tanhf) || __has_builtin(__builtin_amdgcn_tanh_f32)
  return fmaf(0.5f, fast_tanh(0.5f * x), 0.5f);
#else
  return fast_rcp(1.0f + __expf(-x));
#endif
}

__device__ __forceinline__ unsigned f32_e4m3_sw(float f) {
  unsigned u = __float_as_uint(f);
  unsigned s = (u >> 24) & 0x80u;
  float a = fabsf(f);
  if (!(a >= 0.015625f)) {
    int m = (int)(a * 512.0f + 0.5f);
    if (m > 7) m = 7;
    return s | (unsigned)m;
  }
  if (a > 448.0f) return s | 0x7Eu;
  unsigned au = u & 0x7FFFFFFFu;
  unsigned r  = au + 0x0007FFFFu + ((au >> 20) & 1u);
  unsigned e  = (r >> 23) - 120u;
  unsigned m3 = (r >> 20) & 7u;
  if (e > 15u) return s | 0x7Eu;
  return s | (e << 3) | m3;
}

__device__ __forceinline__ unsigned char to_e4m3(float f) {
#if __has_builtin(__builtin_amdgcn_cvt_pk_fp8_f32)
  return (unsigned char)(__builtin_amdgcn_cvt_pk_fp8_f32(f, 0.0f, 0, false) & 0xff);
#else
  return (unsigned char)f32_e4m3_sw(f);
#endif
}

__device__ __forceinline__ v8i lds_load_A(const unsigned char* base, int kc, int lane) {
  const unsigned char* p = base + (lane & 15) * WPAD + kc * 64 + ((lane >> 4) << 4);
  union { v8i v; unsigned long long q[4]; } u;
  u.q[0] = *(const unsigned long long*)(p);
  u.q[1] = *(const unsigned long long*)(p + 8);
  u.q[2] = *(const unsigned long long*)(p + 32);
  u.q[3] = *(const unsigned long long*)(p + 40);
  return u.v;
}

__device__ __forceinline__ v8i lds_load_B(const unsigned char* base, int n0, int kc, int lane) {
  const unsigned char* p = base + (n0 + (lane & 15)) * WPAD + kc * 64 + ((lane >> 4) << 4);
  union { v8i v; u32x4 h[2]; } u;
  u.h[0] = *(const u32x4*)(p);
  u.h[1] = *(const u32x4*)(p + 32);
  return u.v;
}

#define WMMA_FP8(A, B, C) \
  __builtin_amdgcn_wmma_f32_16x16x64_fp8_fp8((A), (B), (short)0, (C), false, false)

__global__ __launch_bounds__(256) void lstm_fused(
    const float* __restrict__ x,
    const float* __restrict__ Wih1, const float* __restrict__ Whh1,
    const float* __restrict__ bih1, const float* __restrict__ bhh1,
    const float* __restrict__ Wih2, const float* __restrict__ Whh2,
    const float* __restrict__ bih2, const float* __restrict__ bhh2,
    const float* __restrict__ W1,   const float* __restrict__ b1,
    const float* __restrict__ W2,   const float* __restrict__ b2,
    float* __restrict__ out)
{
  __shared__ __align__(16) unsigned char smem[SMEM_SZ];

  const int tid   = threadIdx.x;
  const int wv    = tid >> 5;
  const int lane  = tid & 31;
  const int nn    = lane & 15;
  const int hi    = lane >> 4;
  const int mBase = blockIdx.x * MTILE;

  for (int i = tid; i < (2 * 16 * WPAD + 4 * 16 * WPAD) / 4; i += 256)
    ((unsigned int*)(smem + OFF_W1))[i] = 0u;
  __syncthreads();

  for (int i = tid; i < GATES * HID; i += 256) {
    int n = i >> 7, k = i & 127;
    smem[OFF_WHH1 + n * WPAD + k] = to_e4m3(Whh1[i]);
    smem[OFF_WIH2 + n * WPAD + k] = to_e4m3(Wih2[i]);
    smem[OFF_WHH2 + n * WPAD + k] = to_e4m3(Whh2[i]);
  }
  for (int i = tid; i < NOUT * HID; i += 256) {
    int n = i >> 7, k = i & 127;
    smem[OFF_W1 + n * WPAD + k] = to_e4m3(W1[i]);
    smem[OFF_W2 + n * WPAD + k] = to_e4m3(W2[i]);
  }
  float* bs1 = (float*)(smem + OFF_BS1);
  float* bs2 = (float*)(smem + OFF_BS2);
  float* wx1 = (float*)(smem + OFF_WX1);
  float* b1c = (float*)(smem + OFF_B1C);
  float* b2c = (float*)(smem + OFF_B2C);
  float* xst = (float*)(smem + OFF_X);
  for (int i = tid; i < GATES; i += 256) {
    bs1[i] = bih1[i] + bhh1[i];
    bs2[i] = bih2[i] + bhh2[i];
    wx1[i] = Wih1[i];
  }
  if (tid < 16) {
    b1c[tid] = (tid < 8) ? b1[tid] : 0.0f;
    b2c[tid] = (tid < 8) ? b2[tid] : 0.0f;
    xst[tid]      = x[(size_t)(mBase + tid) * TSTEPS];
    xst[16 + tid] = 0.0f;
  }
  __syncthreads();

  const int jn = wv * 16 + nn;
  float bs1r[4], bs2r[4], wxr[4];
#pragma unroll
  for (int q = 0; q < 4; ++q) {
    bs1r[q] = bs1[q * 128 + jn];
    bs2r[q] = bs2[q * 128 + jn];
    wxr[q]  = wx1[q * 128 + jn];
  }
  const float b1v = b1c[nn];
  const float b2v = b2c[nn];

  float c1r[8], c2r[8];
#pragma unroll
  for (int r = 0; r < 8; ++r) { c1r[r] = 0.0f; c2r[r] = 0.0f; }

  v8f zf;
#pragma unroll
  for (int i = 0; i < 8; ++i) zf[i] = 0.0f;

#pragma unroll 1
  for (int t = 0; t < TSTEPS; ++t) {
    const int cur = t & 1, prv = cur ^ 1;
    unsigned char* h1p = smem + OFF_H1 + prv * (16 * WPAD);
    unsigned char* h1c = smem + OFF_H1 + cur * (16 * WPAD);
    unsigned char* h2p = smem + OFF_H2 + prv * (16 * WPAD);
    unsigned char* h2c = smem + OFF_H2 + cur * (16 * WPAD);

    float xv[8];
    {
      const float* xb = xst + cur * 16 + hi * 8;
#pragma unroll
      for (int r = 0; r < 8; ++r) xv[r] = xb[r];
    }

    v8f acc[4] = { zf, zf, zf, zf };
#pragma unroll
    for (int kc = 0; kc < 2; ++kc) {
      v8i a = lds_load_A(h1p, kc, lane);
#pragma unroll
      for (int q = 0; q < 4; ++q) {
        v8i b = lds_load_B(smem + OFF_WHH1, q * 128 + wv * 16, kc, lane);
        acc[q] = WMMA_FP8(a, b, acc[q]);
      }
    }
#pragma unroll
    for (int r = 0; r < 8; ++r) {
      float gi = acc[0][r] + bs1r[0] + xv[r] * wxr[0];
      float gf = acc[1][r] + bs1r[1] + xv[r] * wxr[1];
      float gg = acc[2][r] + bs1r[2] + xv[r] * wxr[2];
      float go = acc[3][r] + bs1r[3] + xv[r] * wxr[3];
      float c  = fast_sigm(gf) * c1r[r] + fast_sigm(gi) * fast_tanh(gg);
      c1r[r]   = c;
      float h  = fast_sigm(go) * fast_tanh(c);
      h1c[(hi * 8 + r) * WPAD + jn] = to_e4m3(h);
    }
    __syncthreads();

#pragma unroll
    for (int q = 0; q < 4; ++q) acc[q] = zf;
#pragma unroll
    for (int kc = 0; kc < 2; ++kc) {
      v8i a = lds_load_A(h1c, kc, lane);
#pragma unroll
      for (int q = 0; q < 4; ++q) {
        v8i b = lds_load_B(smem + OFF_WIH2, q * 128 + wv * 16, kc, lane);
        acc[q] = WMMA_FP8(a, b, acc[q]);
      }
    }
#pragma unroll
    for (int kc = 0; kc < 2; ++kc) {
      v8i a = lds_load_A(h2p, kc, lane);
#pragma unroll
      for (int q = 0; q < 4; ++q) {
        v8i b = lds_load_B(smem + OFF_WHH2, q * 128 + wv * 16, kc, lane);
        acc[q] = WMMA_FP8(a, b, acc[q]);
      }
    }
#pragma unroll
    for (int r = 0; r < 8; ++r) {
      float gi = acc[0][r] + bs2r[0];
      float gf = acc[1][r] + bs2r[1];
      float gg = acc[2][r] + bs2r[2];
      float go = acc[3][r] + bs2r[3];
      float c  = fast_sigm(gf) * c2r[r] + fast_sigm(gi) * fast_tanh(gg);
      c2r[r]   = c;
      float h  = fast_sigm(go) * fast_tanh(c);
      h2c[(hi * 8 + r) * WPAD + jn] = to_e4m3(h);
    }
    if (wv == 7 && lane < 16 && (t + 1) < TSTEPS)
      xst[((t + 1) & 1) * 16 + lane] = x[(size_t)(mBase + lane) * TSTEPS + (t + 1)];
    __syncthreads();

    if (wv == 0) {
      v8f ao = zf;
#pragma unroll
      for (int kc = 0; kc < 2; ++kc) {
        v8i a = lds_load_A(h2c, kc, lane);
        v8i b = lds_load_B(smem + OFF_W1, 0, kc, lane);
        ao = WMMA_FP8(a, b, ao);
      }
      if (nn < 8) {
#pragma unroll
        for (int r = 0; r < 8; ++r) {
          size_t m = (size_t)(mBase + hi * 8 + r);
          out[(size_t)BATCH * NOUT + m * (size_t)TSTEPS * NOUT + (size_t)t * NOUT + nn] =
              ao[r] + b1v;
        }
      }
    }
  }

  if (wv == 0) {
    unsigned char* h2l = smem + OFF_H2 + ((TSTEPS - 1) & 1) * (16 * WPAD);
    v8f ao = zf;
#pragma unroll
    for (int kc = 0; kc < 2; ++kc) {
      v8i a = lds_load_A(h2l, kc, lane);
      v8i b = lds_load_B(smem + OFF_W2, 0, kc, lane);
      ao = WMMA_FP8(a, b, ao);
    }
    if (nn < 8) {
#pragma unroll
      for (int r = 0; r < 8; ++r) {
        size_t m = (size_t)(mBase + hi * 8 + r);
        out[m * NOUT + nn] = ao[r] + b2v;
      }
    }
  }
}

extern "C" void kernel_launch(void* const* d_in, const int* in_sizes, int n_in,
                              void* d_out, int out_size, void* d_ws, size_t ws_size,
                              hipStream_t stream) {
  (void)in_sizes; (void)n_in; (void)out_size; (void)d_ws; (void)ws_size;
  const float* x    = (const float*)d_in[0];
  const float* Wih1 = (const float*)d_in[2];
  const float* Whh1 = (const float*)d_in[3];
  const float* bih1 = (const float*)d_in[4];
  const float* bhh1 = (const float*)d_in[5];
  const float* Wih2 = (const float*)d_in[6];
  const float* Whh2 = (const float*)d_in[7];
  const float* bih2 = (const float*)d_in[8];
  const float* bhh2 = (const float*)d_in[9];
  const float* W1   = (const float*)d_in[10];
  const float* b1   = (const float*)d_in[11];
  const float* W2   = (const float*)d_in[12];
  const float* b2   = (const float*)d_in[13];
  float* out = (float*)d_out;

  dim3 grid(BATCH / MTILE);
  dim3 block(256);
  hipLaunchKernelGGL(lstm_fused, grid, block, 0, stream,
                     x, Wih1, Whh1, bih1, bhh1,
                     Wih2, Whh2, bih2, bhh2,
                     W1, b1, W2, b2, out);
}
